// NSVQ_35356170780841
// MI455X (gfx1250) — matrix operand order repaired and checked by reading, not compiled, not run on hardware
//
#include <hip/hip_runtime.h>
#include <hip/hip_bf16.h>
#include <math.h>
#include <stdint.h>

typedef __attribute__((ext_vector_type(16))) _Float16 v16h;
typedef __attribute__((ext_vector_type(8)))  _Float16 v8h;
typedef __attribute__((ext_vector_type(8)))  float    v8f;

#define K_CODES     1024
#define DIM         64
#define HW          1024          // H*W
#define ROWS_PER_WG 128
#define N_ROWS      32768         // B*H*W
#define CBPAD       72            // codebook LDS row stride in halves (conflict-free b128)
#define XPAD        72            // x-tile LDS row stride in halves
#define NPAD        66            // noise LDS row stride in floats (8B-aligned rows, conflict-free)

union V16U { v16h v; v8h h[2]; };

// ---------- kernel 0: ||c||^2 per codebook row + zero the usage counts ----------
__global__ void nsvq_pre(const float* __restrict__ cb,
                         float* __restrict__ cnorm,
                         int* __restrict__ counts) {
  int c = blockIdx.x * blockDim.x + threadIdx.x;
  if (c < K_CODES) {
    const float* row = cb + (size_t)c * DIM;
    float s = 0.f;
#pragma unroll 8
    for (int d = 0; d < DIM; ++d) { float v = row[d]; s += v * v; }
    cnorm[c]  = s;
    counts[c] = 0;
  }
}

// ---------- kernel 1: fused distances (WMMA f16) + argmin + NSVQ output ----------
__global__ __launch_bounds__(256, 1) void nsvq_main(
    const float* __restrict__ in,      // (B, D, H, W) f32
    const float* __restrict__ cbg,     // (K, D) f32
    const float* __restrict__ noise,   // (N, D) f32
    const float* __restrict__ cnormg,  // (K,) f32
    int* __restrict__ counts,          // (K,) usage histogram
    float* __restrict__ out)           // (B, D, H, W) quantized
{
  extern __shared__ unsigned char smem[];
  _Float16* CB    = (_Float16*)smem;                    // K_CODES * CBPAD halves
  _Float16* XT    = CB + K_CODES * CBPAD;               // ROWS_PER_WG * XPAD halves
  float*    NOISE = (float*)(XT + ROWS_PER_WG * XPAD);  // ROWS_PER_WG * NPAD floats
  float*    CN    = NOISE + ROWS_PER_WG * NPAD;         // K_CODES floats
  float*    NORMR = CN + K_CODES;                       // 128
  float*    NORMB = NORMR + ROWS_PER_WG;                // 128
  float*    RATIO = NORMB + ROWS_PER_WG;                // 128
  int*      BESTI = (int*)(RATIO + ROWS_PER_WG);        // 128

  const int tid = threadIdx.x;
  const int n0  = blockIdx.x * ROWS_PER_WG;             // 128 rows, all in one image
  const int b   = n0 / HW;
  const int hw0 = n0 % HW;
  const float* xbase = in  + (size_t)b * DIM * HW + hw0;
  float*       obase = out + (size_t)b * DIM * HW + hw0;

  if (tid < ROWS_PER_WG) { NORMR[tid] = 0.f; NORMB[tid] = 0.f; }
  for (int idx = tid; idx < K_CODES; idx += 256) CN[idx] = cnormg[idx];
  __syncthreads();

  // noise tile -> LDS via CDNA5 async global->LDS DMA (8B chunks, coalesced, ASYNCcnt)
  for (int idx = tid; idx < ROWS_PER_WG * (DIM / 2); idx += 256) {
    int i = idx >> 5, c = idx & 31;                       // 32 x 8B chunks per row
    uint64_t gaddr = (uint64_t)(uintptr_t)(noise + (size_t)(n0 + i) * DIM + c * 2);
    uint32_t laddr = (uint32_t)(uintptr_t)(NOISE + i * NPAD + c * 2);
    asm volatile("global_load_async_to_lds_b64 %0, %1, off"
                 :: "v"(laddr), "v"(gaddr) : "memory");
  }

  // codebook f32 -> f16 LDS (coalesced over d)
  for (int idx = tid; idx < K_CODES * DIM; idx += 256) {
    int c = idx >> 6, d = idx & 63;
    CB[c * CBPAD + d] = (_Float16)cbg[idx];
  }
  // x tile -> f16 LDS (i fast-moving => contiguous NCHW reads)
  for (int idx = tid; idx < DIM * ROWS_PER_WG; idx += 256) {
    int d = idx >> 7, i = idx & 127;
    XT[i * XPAD + d] = (_Float16)xbase[d * HW + i];
  }
  asm volatile("s_wait_asynccnt 0" ::: "memory");
  __syncthreads();

  // ---- per-wave WMMA distance scan + running argmin ----
  const int lane = tid & 31;
  const int wv   = tid >> 5;          // 8 waves, 16 rows each
  const int m16  = lane & 15;
  const int g    = (lane >> 4) & 1;

  // A fragments (16x32 f16 layout: lanes 0-15 K 0..7/16..23, lanes 16-31 K 8..15/24..31)
  V16U a0, a1;
  {
    const _Float16* xr = XT + (wv * 16 + m16) * XPAD;
    a0.h[0] = *(const v8h*)(xr + g * 8);
    a0.h[1] = *(const v8h*)(xr + 16 + g * 8);
    a1.h[0] = *(const v8h*)(xr + 32 + g * 8);
    a1.h[1] = *(const v8h*)(xr + 48 + g * 8);
  }

  float mind[8];
  int   mini[8];
#pragma unroll
  for (int r = 0; r < 8; ++r) { mind[r] = 3.4e38f; mini[r] = 0; }

#pragma unroll 2
  for (int kk = 0; kk < K_CODES / 16; ++kk) {
    const int c = kk * 16 + m16;                 // this lane's codebook column
    const _Float16* cr = CB + c * CBPAD;
    // B fragments (32x16 f16 layout: lanes 0-15 K 0..7/16..23, lanes 16-31 K 8..15/24..31)
    V16U b0, b1;
    b0.h[0] = *(const v8h*)(cr + g * 8);
    b0.h[1] = *(const v8h*)(cr + 16 + g * 8);
    b1.h[0] = *(const v8h*)(cr + 32 + g * 8);
    b1.h[1] = *(const v8h*)(cr + 48 + g * 8);

    v8f acc = {};
    acc = __builtin_amdgcn_wmma_f32_16x16x32_f16(false, a0.v, false, b0.v,
                                                 (short)0, acc, false, false);
    acc = __builtin_amdgcn_wmma_f32_16x16x32_f16(false, a1.v, false, b1.v,
                                                 (short)0, acc, false, false);
    const float cn2 = CN[c];
#pragma unroll
    for (int r = 0; r < 8; ++r) {
      float dist = cn2 - 2.0f * acc[r];          // ||x||^2 dropped (row-constant)
      if (dist < mind[r]) { mind[r] = dist; mini[r] = c; }  // c increasing -> first-min kept
    }
  }

  // reduce argmin across the 16 column lanes (rows live in one half-wave)
#pragma unroll
  for (int r = 0; r < 8; ++r) {
    float d = mind[r]; int i = mini[r];
#pragma unroll
    for (int off = 8; off >= 1; off >>= 1) {
      float od = __shfl_xor(d, off, 32);
      int   oi = __shfl_xor(i, off, 32);
      if (od < d || (od == d && oi < i)) { d = od; i = oi; }
    }
    if (m16 == 0) BESTI[wv * 16 + g * 8 + r] = i;
  }
  __syncthreads();

  // ---- exact f32 ||x - c_best||^2 and ||noise||^2 per row ----
  for (int idx = tid; idx < DIM * ROWS_PER_WG; idx += 256) {
    int d = idx >> 7, i = idx & 127;
    float xv = xbase[d * HW + i];                 // contiguous per instruction
    float cv = cbg[(size_t)BESTI[i] * DIM + d];   // gather, hot in L2
    float df = xv - cv;
    float nv = NOISE[i * NPAD + d];               // bank (2i+d)%64 -> conflict-free
    atomicAdd(&NORMB[i], df * df);
    atomicAdd(&NORMR[i], nv * nv);
  }
  __syncthreads();

  if (tid < ROWS_PER_WG) {
    RATIO[tid] = sqrtf(NORMB[tid]) / sqrtf(NORMR[tid]) + 1e-12f;
    atomicAdd(&counts[BESTI[tid]], 1);
  }
  __syncthreads();

  // ---- quantized = x + ratio * noise, NCHW-coalesced ----
  for (int idx = tid; idx < DIM * ROWS_PER_WG; idx += 256) {
    int d = idx >> 7, i = idx & 127;
    float xv = xbase[d * HW + i];
    obase[d * HW + i] = xv + RATIO[i] * NOISE[i * NPAD + d];
  }
}

// ---------- kernel 2: perplexity from the usage histogram ----------
__global__ void nsvq_perplexity(const int* __restrict__ counts,
                                float* __restrict__ outp) {
  __shared__ float red[256];
  float s = 0.f;
  for (int c = threadIdx.x; c < K_CODES; c += 256) {
    float p = (float)counts[c] * (1.0f / (float)N_ROWS);
    s += p * logf(p + 1e-10f);
  }
  red[threadIdx.x] = s;
  __syncthreads();
  for (int off = 128; off > 0; off >>= 1) {
    if (threadIdx.x < off) red[threadIdx.x] += red[threadIdx.x + off];
    __syncthreads();
  }
  if (threadIdx.x == 0) *outp = expf(-red[0]);
}

extern "C" void kernel_launch(void* const* d_in, const int* in_sizes, int n_in,
                              void* d_out, int out_size, void* d_ws, size_t ws_size,
                              hipStream_t stream) {
  const float* in    = (const float*)d_in[0];   // (32,64,32,32) f32
  const float* cb    = (const float*)d_in[1];   // (1024,64) f32
  const float* noise = (const float*)d_in[2];   // (32768,64) f32
  float* out = (float*)d_out;                   // 32*64*32*32 quantized + 1 perplexity

  int*   counts = (int*)d_ws;
  float* cnorm  = (float*)((char*)d_ws + K_CODES * sizeof(int));

  nsvq_pre<<<(K_CODES + 255) / 256, 256, 0, stream>>>(cb, cnorm, counts);

  const size_t smem =
      (size_t)(K_CODES * CBPAD + ROWS_PER_WG * XPAD) * sizeof(_Float16) +
      (size_t)(ROWS_PER_WG * NPAD + K_CODES + 3 * ROWS_PER_WG) * sizeof(float) +
      (size_t)ROWS_PER_WG * sizeof(int);

  nsvq_main<<<N_ROWS / ROWS_PER_WG, 256, smem, stream>>>(in, cb, noise, cnorm,
                                                         counts, out);

  nsvq_perplexity<<<1, 256, 0, stream>>>(counts, out + (size_t)N_ROWS * DIM);
}
